// SO3LinearV2_12163347382325
// MI455X (gfx1250) — matrix operand order repaired and checked by reading, not compiled, not run on hardware
//
#include <hip/hip_runtime.h>

// ---------------------------------------------------------------------------
// SO3 equivariant linear (equiformer-style) for MI455X / gfx1250.
//   out[b,m,n] = sum_k emb[b,m,k] * W[l(m)][k][n]   (+ bias[n] iff m==0)
//
// Roofline: 128.5 GFLOP vs ~2.0 GB mandatory HBM traffic -> ~86 us memory
// bound at 23.3 TB/s. Weight (1.75 MB) is L2-resident. Compute path:
// f32 -> (bf16 hi, bf16 lo) split, 3x V_WMMA_F32_16X16X32_BF16 per
// 16x16x32 tile (~tf32-class accuracy) keeps matrix-issue time well under
// the memory floor, unlike the K=4 f32 WMMA.
//
// Per-block: m fixed (grid.y), 128 b-rows (grid.x), 8 waves x 16 rows,
// each wave owns all 256 output columns (16 n-tiles, 128 acc VGPRs) so the
// embedding is streamed from HBM exactly once. Weight k-chunks are
// transposed + hi/lo-split into LDS cooperatively per k-step; A-fragment
// global loads are issued before the fill barrier to hide HBM latency.
// ---------------------------------------------------------------------------

typedef __bf16 bf16_t;
typedef bf16_t v16bf __attribute__((ext_vector_type(16)));
typedef float  v8f   __attribute__((ext_vector_type(8)));

#define SO3_NUM_M 49
#define SO3_IN_F  256
#define SO3_OUT_F 256
#define SO3_BK    32     // k-chunk per step (WMMA K)
#define SO3_KROW  48     // padded LDS row stride in bf16 elems (96B: 32B-aligned frags)
#define SO3_BM    128    // b-rows per block (8 waves x 16)
#define SO3_NT    16     // 16-wide n-tiles per wave (16*16 = 256 cols)

__global__ __launch_bounds__(256)
void so3_linear_wmma(const float* __restrict__ emb,
                     const float* __restrict__ w,
                     const float* __restrict__ bias,
                     const int*   __restrict__ expand_index,
                     float* __restrict__ out,
                     int Bn)
{
    // hi / lo bf16 copies of the current 32(K) x 256(N) weight chunk,
    // stored [n][kk], kk contiguous, padded stride for aligned b128 frags.
    __shared__ bf16_t sBh[SO3_OUT_F * SO3_KROW];
    __shared__ bf16_t sBl[SO3_OUT_F * SO3_KROW];

    const int m    = blockIdx.y;
    const int l    = expand_index[m];
    const int tid  = threadIdx.x;
    const int wave = tid >> 5;
    const int lane = tid & 31;
    const int half = lane >> 4;   // 16-lane group
    const int mr   = lane & 15;   // M (for A) / N (for B,D) index within tile

    const int rowbase = blockIdx.x * SO3_BM + wave * 16;
    // Clamp for loads; stores are guarded. (Bn % 16 == 0 keeps tiles uniform.)
    const int lrow = min(rowbase + mr, Bn - 1);
    const float* __restrict__ ap = emb + ((size_t)lrow * SO3_NUM_M + m) * SO3_IN_F;
    const float* __restrict__ wl = w + (size_t)l * SO3_IN_F * SO3_OUT_F;

    v8f acc[SO3_NT];
    #pragma unroll
    for (int i = 0; i < SO3_NT; ++i)
        #pragma unroll
        for (int e = 0; e < 8; ++e) acc[i][e] = 0.0f;

    for (int kc = 0; kc < SO3_IN_F; kc += SO3_BK) {
        // ---- A-fragment global loads FIRST (no LDS dependency): overlap
        // HBM latency with the weight fill + barriers below.
        // A layout (16x32 bf16): lane = row M=mr, K in [kb,kb+16).
        const int kb = kc + half * 16;
        float4 f0 = *(const float4*)(ap + kb);
        float4 f1 = *(const float4*)(ap + kb + 4);
        float4 f2 = *(const float4*)(ap + kb + 8);
        float4 f3 = *(const float4*)(ap + kb + 12);

        // Warm the L2->WGP path for the next weight chunk while we compute.
        if (kc + SO3_BK < SO3_IN_F) {
            __builtin_prefetch(wl + (size_t)(kc + SO3_BK) * SO3_OUT_F + tid, 0, 0);
        }

        __syncthreads();  // previous iteration's LDS readers done

        // ---- cooperative LDS fill: thread `tid` owns output column n = tid.
        // Reads W[l][kc+kk][n] (n-contiguous -> coalesced), splits f32 into
        // bf16 hi + bf16 lo, writes transposed rows [n][kk].
        {
            const int n = tid;
            const float* __restrict__ wp = wl + (size_t)kc * SO3_OUT_F + n;
            #pragma unroll
            for (int kk = 0; kk < SO3_BK; ++kk) {
                float  x   = wp[(size_t)kk * SO3_OUT_F];
                bf16_t h   = (bf16_t)x;
                float  rem = x - (float)h;
                sBh[n * SO3_KROW + kk] = h;
                sBl[n * SO3_KROW + kk] = (bf16_t)rem;
            }
        }
        __syncthreads();

        // ---- split A into hi/lo bf16 fragments.
        float af[16] = { f0.x, f0.y, f0.z, f0.w,  f1.x, f1.y, f1.z, f1.w,
                         f2.x, f2.y, f2.z, f2.w,  f3.x, f3.y, f3.z, f3.w };
        v16bf ah, al;
        #pragma unroll
        for (int e = 0; e < 16; ++e) {
            bf16_t h = (bf16_t)af[e];
            ah[e] = h;
            al[e] = (bf16_t)(af[e] - (float)h);
        }

        // ---- 16 n-tiles, 3 WMMAs each (hh, hl, lh).
        #pragma unroll
        for (int nt = 0; nt < SO3_NT; ++nt) {
            const int n = nt * 16 + mr;   // B layout: lane carries column N
            const v16bf bh = *(const v16bf*)(&sBh[n * SO3_KROW + half * 16]);
            const v16bf bl = *(const v16bf*)(&sBl[n * SO3_KROW + half * 16]);
            acc[nt] = __builtin_amdgcn_wmma_f32_16x16x32_bf16(
                false, ah, false, bh, (short)0, acc[nt], false, false);
            acc[nt] = __builtin_amdgcn_wmma_f32_16x16x32_bf16(
                false, ah, false, bl, (short)0, acc[nt], false, false);
            acc[nt] = __builtin_amdgcn_wmma_f32_16x16x32_bf16(
                false, al, false, bh, (short)0, acc[nt], false, false);
        }
    }

    // ---- epilogue: D layout is lane -> N = mr, reg r -> M = r + 8*half.
    const bool addb = (m == 0);
    #pragma unroll
    for (int nt = 0; nt < SO3_NT; ++nt) {
        const int n  = nt * 16 + mr;
        const float bv = addb ? bias[n] : 0.0f;
        #pragma unroll
        for (int r = 0; r < 8; ++r) {
            const int orow = rowbase + r + 8 * half;
            if (orow < Bn) {
                out[((size_t)orow * SO3_NUM_M + m) * SO3_OUT_F + n] = acc[nt][r] + bv;
            }
        }
    }
}

extern "C" void kernel_launch(void* const* d_in, const int* in_sizes, int n_in,
                              void* d_out, int out_size, void* d_ws, size_t ws_size,
                              hipStream_t stream) {
    const float* emb  = (const float*)d_in[0];
    const float* w    = (const float*)d_in[1];
    const float* bias = (const float*)d_in[2];
    const int*   ei   = (const int*)d_in[3];
    float* out = (float*)d_out;

    const int Bn = in_sizes[0] / (SO3_NUM_M * SO3_IN_F);  // 20000
    dim3 grid((Bn + SO3_BM - 1) / SO3_BM, SO3_NUM_M);
    so3_linear_wmma<<<grid, dim3(256), 0, stream>>>(emb, w, bias, ei, out, Bn);
}
